// QuantumInspiredAttention_14654428414323
// MI455X (gfx1250) — matrix operand order repaired and checked by reading, not compiled, not run on hardware
//
#include <hip/hip_runtime.h>
#include <hip/hip_bf16.h>
#include <math.h>

// ---------- CDNA5 WMMA types ----------
typedef __attribute__((ext_vector_type(16))) __bf16          v16bf;
typedef __attribute__((ext_vector_type(8)))  float           v8f;
typedef __attribute__((ext_vector_type(8)))  unsigned short  us8;

union Frag {
  v16bf v;
  us8   u8[2];
  unsigned short h[16];
};

__device__ __forceinline__ v8f wmma_bf16(const Frag& a, const Frag& b, v8f c) {
  // v_wmma_f32_16x16x32_bf16 : D = A(16x32) * B(32x16) + C(16x16 f32)
  return __builtin_amdgcn_wmma_f32_16x16x32_bf16(
      /*neg_a=*/false, a.v, /*neg_b=*/false, b.v,
      /*c_mod=*/(short)0, c, /*reuse_a=*/false, /*reuse_b=*/false);
}

__device__ __forceinline__ unsigned short f2bf(float f) {
  unsigned int u = __float_as_uint(f);
  u += 0x7FFFu + ((u >> 16) & 1u);      // round-to-nearest-even
  return (unsigned short)(u >> 16);
}

static constexpr int Bsz = 2, SEQ = 2048, DIM = 512, H = 8, HD = 64;
static constexpr int M_ROWS = Bsz * SEQ;   // 4096
static constexpr int E3     = 3 * DIM;     // 1536

// ---------- kernel 0: f32 -> bf16 convert ----------
__global__ void cvt_bf16(const float* __restrict__ src,
                         unsigned short* __restrict__ dst, int n) {
  int i = blockIdx.x * blockDim.x + threadIdx.x;
  if (i < n) dst[i] = f2bf(src[i]);
}

// ---------- kernel 1: QKV projection (4096 x 1536 x 512) ----------
// Block tile 128x64, 8 waves of 32x32 (4 accumulators each -> 2 b128 loads/WMMA).
// Epilogue adds bias, scales Q by hd^-0.5, scatters into head-major Q/K and V^T.
__global__ __launch_bounds__(256) void qkv_gemm(
    const unsigned short* __restrict__ xb,    // [4096][512] bf16
    const unsigned short* __restrict__ wb,    // [1536][512] bf16 (qkv_w)
    const float* __restrict__ qkv_b,          // [1536]
    unsigned short* __restrict__ Qb,          // [B,H,N,hd]
    unsigned short* __restrict__ Kb,          // [B,H,N,hd]
    unsigned short* __restrict__ Vtb)         // [B,H,hd,N]
{
  const int lane = threadIdx.x & 31;
  const int wave = threadIdx.x >> 5;
  const int wm = wave & 3, wn = wave >> 2;
  const int m0 = blockIdx.x * 128 + wm * 32;
  const int n0 = blockIdx.y * 64 + wn * 32;

  const int lm = lane & 15;
  const int hi = lane >> 4;
  const int kgrpA  = hi ? 8  : 0;   // A-frag K-group per ISA 16-bit layout
  const int kbaseB = hi ? 8 : 0;    // B-frag K-base  per ISA 16-bit layout

  const unsigned short* arow0 = xb + (size_t)(m0 + lm) * DIM;
  const unsigned short* arow1 = xb + (size_t)(m0 + 16 + lm) * DIM;
  const unsigned short* brow0 = wb + (size_t)(n0 + lm) * DIM;
  const unsigned short* brow1 = wb + (size_t)(n0 + 16 + lm) * DIM;

  v8f acc[2][2] = {};
  for (int k = 0; k < DIM; k += 32) {
    Frag a[2], b[2];
    a[0].u8[0] = *(const us8*)(arow0 + k + kgrpA);
    a[0].u8[1] = *(const us8*)(arow0 + k + kgrpA + 16);
    a[1].u8[0] = *(const us8*)(arow1 + k + kgrpA);
    a[1].u8[1] = *(const us8*)(arow1 + k + kgrpA + 16);
    b[0].u8[0] = *(const us8*)(brow0 + k + kbaseB);
    b[0].u8[1] = *(const us8*)(brow0 + k + kbaseB + 16);
    b[1].u8[0] = *(const us8*)(brow1 + k + kbaseB);
    b[1].u8[1] = *(const us8*)(brow1 + k + kbaseB + 16);
    __builtin_prefetch(arow0 + k + 128, 0, 0);   // global_prefetch_b8 (speculative)
    __builtin_prefetch(brow0 + k + 128, 0, 0);
    acc[0][0] = wmma_bf16(a[0], b[0], acc[0][0]);
    acc[0][1] = wmma_bf16(a[0], b[1], acc[0][1]);
    acc[1][0] = wmma_bf16(a[1], b[0], acc[1][0]);
    acc[1][1] = wmma_bf16(a[1], b[1], acc[1][1]);
  }

  for (int mt = 0; mt < 2; ++mt) {
    for (int nt = 0; nt < 2; ++nt) {
      const int e = n0 + nt * 16 + lm;
      const float bias = qkv_b[e];
      const int which = e / DIM;
      const int d = e % DIM;
      const int hh = d >> 6, hd_i = d & 63;
#pragma unroll
      for (int r = 0; r < 8; ++r) {
        const int m = m0 + mt * 16 + r + (hi ? 8 : 0);
        const int bb = m >> 11, n = m & 2047;
        const float v = acc[mt][nt][r] + bias;
        if (which == 0) {
          Qb[((size_t)(bb * H + hh) * SEQ + n) * HD + hd_i] = f2bf(v * 0.125f); // hd^-0.5
        } else if (which == 1) {
          Kb[((size_t)(bb * H + hh) * SEQ + n) * HD + hd_i] = f2bf(v);
        } else {
          Vtb[((size_t)(bb * H + hh) * HD + hd_i) * SEQ + n] = f2bf(v);
        }
      }
    }
  }
}

// ---------- kernel 2: attention: S = QK^T -> l2norm -> softmax -> O = P V ----------
// One WG per (b, h, 16-query block). S (16x2048 f32) + P (16x2048 bf16) in LDS (192KB).
__global__ __launch_bounds__(256) void attn_kernel(
    const unsigned short* __restrict__ Qb,
    const unsigned short* __restrict__ Kb,
    const unsigned short* __restrict__ Vtb,
    unsigned short* __restrict__ AOb)          // [4096][512] bf16 (b,n,h,hd)
{
  extern __shared__ char smem[];
  float*          S = (float*)smem;                                   // [16][2048]
  unsigned short* P = (unsigned short*)(smem + (size_t)16 * SEQ * 4); // [16][2048]

  const int lane = threadIdx.x & 31;
  const int wave = threadIdx.x >> 5;
  const int rowblk = blockIdx.x;   // 0..127
  const int h = blockIdx.y;
  const int b = blockIdx.z;
  const int bh = b * H + h;

  const int lm = lane & 15;
  const int hi = lane >> 4;
  const int kgrpA  = hi ? 8  : 0;
  const int kbaseB = hi ? 8 : 0;

  // ---- phase 1: score tiles, Q(16x64) x K^T(64x16) per tile, A frags register-resident ----
  const unsigned short* qrow = Qb + ((size_t)bh * SEQ + rowblk * 16 + lm) * HD;
  Frag a0, a1;
  a0.u8[0] = *(const us8*)(qrow + kgrpA);
  a0.u8[1] = *(const us8*)(qrow + kgrpA + 16);
  a1.u8[0] = *(const us8*)(qrow + 32 + kgrpA);
  a1.u8[1] = *(const us8*)(qrow + 32 + kgrpA + 16);

  for (int t = wave; t < SEQ / 16; t += 8) {
    const unsigned short* krow = Kb + ((size_t)bh * SEQ + t * 16 + lm) * HD;
    Frag b0, b1;
    b0.u8[0] = *(const us8*)(krow + kbaseB);
    b0.u8[1] = *(const us8*)(krow + kbaseB + 16);
    b1.u8[0] = *(const us8*)(krow + 32 + kbaseB);
    b1.u8[1] = *(const us8*)(krow + 32 + kbaseB + 16);
    __builtin_prefetch(krow + 8 * 16 * HD, 0, 0);  // next tile for this wave
    v8f acc = {};
    acc = wmma_bf16(a0, b0, acc);
    acc = wmma_bf16(a1, b1, acc);
#pragma unroll
    for (int r = 0; r < 8; ++r)
      S[(size_t)(r + (hi ? 8 : 0)) * SEQ + t * 16 + lm] = acc[r];
  }
  __syncthreads();

  // ---- phase 2: per-row L2 norm, softmax, bf16 quantize (wave32 reductions) ----
  for (int row = wave; row < 16; row += 8) {
    float* Srow = S + (size_t)row * SEQ;
    float sumsq = 0.f, mx = -1e30f;
    for (int c = lane; c < SEQ; c += 32) {
      const float s = Srow[c];
      sumsq += s * s;
      mx = fmaxf(mx, s);
    }
#pragma unroll
    for (int off = 16; off > 0; off >>= 1) {
      sumsq += __shfl_xor(sumsq, off, 32);
      mx = fmaxf(mx, __shfl_xor(mx, off, 32));
    }
    const float invc = 1.f / (sqrtf(sumsq) + 1e-6f);
    float sum = 0.f;
    for (int c = lane; c < SEQ; c += 32) {
      const float p = __expf((Srow[c] - mx) * invc);  // softmax(s/c), c > 0
      Srow[c] = p;
      sum += p;
    }
#pragma unroll
    for (int off = 16; off > 0; off >>= 1) sum += __shfl_xor(sum, off, 32);
    const float invs = 1.f / sum;
    for (int c = lane; c < SEQ; c += 32)
      P[(size_t)row * SEQ + c] = f2bf(Srow[c] * invs);
  }
  __syncthreads();

  // ---- phase 3: O = P(16x2048) x V(2048x64), all 8 waves: 4 hd-tiles x 2 K-halves ----
  // f32 partials land in LDS (reusing dead S buffer), combined after a barrier.
  float* Ored = S;                       // [8][8][32] f32 = 8KB
  {
    const int tile = wave & 3;           // hd column tile
    const int half = wave >> 2;          // K half: [0,1024) or [1024,2048)
    const int n0 = tile * 16;
    const unsigned short* vt  = Vtb + ((size_t)bh * HD + n0 + lm) * SEQ;
    const unsigned short* prw = P + (size_t)lm * SEQ;  // LDS row
    v8f acc = {};
    const int k0 = half * (SEQ / 2), k1 = k0 + SEQ / 2;
    for (int k = k0; k < k1; k += 32) {
      Frag a, bf;
      a.u8[0]  = *(const us8*)(prw + k + kgrpA);       // ds_load_b128
      a.u8[1]  = *(const us8*)(prw + k + kgrpA + 16);
      bf.u8[0] = *(const us8*)(vt + k + kbaseB);
      bf.u8[1] = *(const us8*)(vt + k + kbaseB + 16);
      acc = wmma_bf16(a, bf, acc);
    }
#pragma unroll
    for (int r = 0; r < 8; ++r)
      Ored[(size_t)wave * 256 + r * 32 + lane] = acc[r];
  }
  __syncthreads();
  if (wave < 4) {
#pragma unroll
    for (int r = 0; r < 8; ++r) {
      const float v = Ored[(size_t)wave * 256 + r * 32 + lane] +
                      Ored[(size_t)(wave + 4) * 256 + r * 32 + lane];
      const int n_out = rowblk * 16 + r + (hi ? 8 : 0);
      const int col = h * HD + wave * 16 + lm;
      AOb[((size_t)(b * SEQ + n_out)) * DIM + col] = f2bf(v);
    }
  }
}

// ---------- kernel 3: output projection (4096 x 512 x 512) + bias -> f32 ----------
// Block tile 128x64, 8 waves of 32x32.
__global__ __launch_bounds__(256) void proj_gemm(
    const unsigned short* __restrict__ AOb,   // [4096][512] bf16
    const unsigned short* __restrict__ wb,    // [512][512]  bf16 (proj_w)
    const float* __restrict__ proj_b,         // [512]
    float* __restrict__ out)                  // [4096][512] f32
{
  const int lane = threadIdx.x & 31;
  const int wave = threadIdx.x >> 5;
  const int wm = wave & 3, wn = wave >> 2;
  const int m0 = blockIdx.x * 128 + wm * 32;
  const int n0 = blockIdx.y * 64 + wn * 32;

  const int lm = lane & 15;
  const int hi = lane >> 4;
  const int kgrpA  = hi ? 8  : 0;
  const int kbaseB = hi ? 8 : 0;

  const unsigned short* arow0 = AOb + (size_t)(m0 + lm) * DIM;
  const unsigned short* arow1 = AOb + (size_t)(m0 + 16 + lm) * DIM;
  const unsigned short* brow0 = wb  + (size_t)(n0 + lm) * DIM;
  const unsigned short* brow1 = wb  + (size_t)(n0 + 16 + lm) * DIM;

  v8f acc[2][2] = {};
  for (int k = 0; k < DIM; k += 32) {
    Frag a[2], b[2];
    a[0].u8[0] = *(const us8*)(arow0 + k + kgrpA);
    a[0].u8[1] = *(const us8*)(arow0 + k + kgrpA + 16);
    a[1].u8[0] = *(const us8*)(arow1 + k + kgrpA);
    a[1].u8[1] = *(const us8*)(arow1 + k + kgrpA + 16);
    b[0].u8[0] = *(const us8*)(brow0 + k + kbaseB);
    b[0].u8[1] = *(const us8*)(brow0 + k + kbaseB + 16);
    b[1].u8[0] = *(const us8*)(brow1 + k + kbaseB);
    b[1].u8[1] = *(const us8*)(brow1 + k + kbaseB + 16);
    __builtin_prefetch(arow0 + k + 128, 0, 0);
    __builtin_prefetch(brow0 + k + 128, 0, 0);
    acc[0][0] = wmma_bf16(a[0], b[0], acc[0][0]);
    acc[0][1] = wmma_bf16(a[0], b[1], acc[0][1]);
    acc[1][0] = wmma_bf16(a[1], b[0], acc[1][0]);
    acc[1][1] = wmma_bf16(a[1], b[1], acc[1][1]);
  }

  for (int mt = 0; mt < 2; ++mt) {
    for (int nt = 0; nt < 2; ++nt) {
      const int e = n0 + nt * 16 + lm;
      const float bias = proj_b[e];
#pragma unroll
      for (int r = 0; r < 8; ++r) {
        const int m = m0 + mt * 16 + r + (hi ? 8 : 0);
        out[(size_t)m * DIM + e] = acc[mt][nt][r] + bias;
      }
    }
  }
}

// ---------- launcher ----------
extern "C" void kernel_launch(void* const* d_in, const int* in_sizes, int n_in,
                              void* d_out, int out_size, void* d_ws, size_t ws_size,
                              hipStream_t stream) {
  (void)in_sizes; (void)n_in; (void)out_size; (void)ws_size;
  const float* x      = (const float*)d_in[0];
  const float* qkv_w  = (const float*)d_in[1];
  const float* qkv_b  = (const float*)d_in[2];
  const float* proj_w = (const float*)d_in[3];
  const float* proj_b = (const float*)d_in[4];
  float* out = (float*)d_out;

  unsigned short* ws = (unsigned short*)d_ws;
  size_t off = 0;
  unsigned short* xb    = ws + off; off += (size_t)M_ROWS * DIM;  // 4 MB
  unsigned short* wqkvb = ws + off; off += (size_t)E3 * DIM;      // 1.5 MB
  unsigned short* wprjb = ws + off; off += (size_t)DIM * DIM;     // 0.5 MB
  unsigned short* Qb    = ws + off; off += (size_t)M_ROWS * DIM;  // 4 MB
  unsigned short* Kb    = ws + off; off += (size_t)M_ROWS * DIM;  // 4 MB
  unsigned short* Vtb   = ws + off; off += (size_t)M_ROWS * DIM;  // 4 MB
  unsigned short* AOb   = ws + off; off += (size_t)M_ROWS * DIM;  // 4 MB

  cvt_bf16<<<(M_ROWS * DIM + 255) / 256, 256, 0, stream>>>(x, xb, M_ROWS * DIM);
  cvt_bf16<<<(E3 * DIM + 255) / 256, 256, 0, stream>>>(qkv_w, wqkvb, E3 * DIM);
  cvt_bf16<<<(DIM * DIM + 255) / 256, 256, 0, stream>>>(proj_w, wprjb, DIM * DIM);

  qkv_gemm<<<dim3(M_ROWS / 128, E3 / 64), 256, 0, stream>>>(xb, wqkvb, qkv_b, Qb, Kb, Vtb);

  const int smem = 16 * SEQ * 4 + 16 * SEQ * 2;  // 192 KB (CDNA5 WGP has 320 KB LDS)
  hipFuncSetAttribute((const void*)attn_kernel,
                      hipFuncAttributeMaxDynamicSharedMemorySize, smem);
  attn_kernel<<<dim3(SEQ / 16, H, Bsz), 256, smem, stream>>>(Qb, Kb, Vtb, AOb);

  proj_gemm<<<dim3(M_ROWS / 128, DIM / 64), 256, 0, stream>>>(AOb, wprjb, proj_b, out);
}
